// ParallelLinearQuantized_19292993094040
// MI455X (gfx1250) — matrix operand order repaired and checked by reading, not compiled, not run on hardware
//
#include <hip/hip_runtime.h>

#define IN_F   4096
#define OUT_F  14336
#define GS     128
#define M_TOT  256
#define BN     128
#define KSLICES 4
#define K_PER  (IN_F / KSLICES)        // 1024
#define GROUPS_PER_SLICE (K_PER / GS)  // 8
#define KSTEP  32
#define NSTEPS (K_PER / KSTEP)         // 32

typedef _Float16 v16h __attribute__((ext_vector_type(16)));
typedef _Float16 v8h  __attribute__((ext_vector_type(8)));
typedef float    v8f  __attribute__((ext_vector_type(8)));
typedef float    v4f  __attribute__((ext_vector_type(4)));

// out[m][n] = bias[n]  (also overwrites the 0xAA poison deterministically)
__global__ __launch_bounds__(256) void qlinear_init_bias(const float* __restrict__ bias,
                                                         float* __restrict__ out) {
  unsigned i = blockIdx.x * 256u + threadIdx.x;      // exact multiple: 256*14336
  out[i] = bias[i % OUT_F];
}

// Issue the 16 strided weight loads for one K-step (perfectly coalesced
// across lanes: 16 consecutive dwords per half-wave) + prefetch next rows.
__device__ __forceinline__ void load_q16(const int* __restrict__ qk, int k0,
                                         int half, int wn, int q[16]) {
  const int* qp = qk + (size_t)(k0 + 16 * half) * OUT_F + wn;
  __builtin_prefetch(qp + (size_t)KSTEP * OUT_F, 0, 1);   // global_prefetch_b8
#pragma unroll
  for (int i = 0; i < 16; ++i) q[i] = qp[(size_t)i * OUT_F];
}

__device__ __forceinline__ void dequant16(const int q[16], float sc, float zb,
                                          v8h& lo, v8h& hi) {
#pragma unroll
  for (int i = 0; i < 8; ++i) lo[i] = (_Float16)fmaf((float)q[i], sc, zb);
#pragma unroll
  for (int i = 0; i < 8; ++i) hi[i] = (_Float16)fmaf((float)q[i + 8], sc, zb);
}

// Block: 256 threads = 8 waves. Block tile: all 256 M rows x 128 N cols over a
// 1024-deep K slice. Wave w owns M rows [32w,32w+32) (2 m-tiles) and all 8
// n-tiles -> 16 f32 16x16 accumulators. Weights stream once from HBM,
// dequantized int32 -> f16 in registers, software-pipelined one K-step ahead
// through a double-buffered fragment-ordered LDS tile.
__global__ __launch_bounds__(256) void qlinear_wmma(const float* __restrict__ x,
                                                    const int*   __restrict__ qk,
                                                    const float* __restrict__ qs,
                                                    const float* __restrict__ qb,
                                                    float*       __restrict__ out) {
  // Fragment order: slot t holds 16 f16 (lane t%32 of n-tile t/32), padded to
  // 24 halves (48 B) per slot -> conflict-free ds_load_b128 pairs. x2 buffers.
  __shared__ __align__(16) _Float16 Bsh[2][256 * 24];   // 24 KB

  const int tid  = threadIdx.x;
  const int wave = tid >> 5;
  const int lane = tid & 31;
  const int half = lane >> 4;
  const int l16  = lane & 15;

  const int n0     = blockIdx.x * BN;
  const int kslice = blockIdx.y;
  const int kbase  = kslice * K_PER;

  // Writer role: thread tid dequantizes column wn, k rows [16*half, 16*half+16)
  const int wn = n0 + wave * 16 + l16;   // LDS slot == tid by construction

  v8f acc[2][8];
  {
    const v8f z = {0.f, 0.f, 0.f, 0.f, 0.f, 0.f, 0.f, 0.f};
#pragma unroll
    for (int i = 0; i < 2; ++i)
#pragma unroll
      for (int j = 0; j < 8; ++j) acc[i][j] = z;
  }

  // ---- prologue: load + dequantize K-step 0 ----
  int  qreg[16];
  v8h  blo, bhi;
  {
    load_q16(qk, kbase, half, wn, qreg);
    const int gidx = kslice * GROUPS_PER_SLICE;
    const float sc = qs[(size_t)gidx * OUT_F + wn];
    const float zb = qb[(size_t)gidx * OUT_F + wn];
    dequant16(qreg, sc, zb, blo, bhi);
  }

  for (int ks = 0; ks < NSTEPS; ++ks) {
    const int p  = ks & 1;
    const int k0 = kbase + ks * KSTEP;

    // ---- publish current dequantized fragment to LDS buffer p ----
    {
      _Float16* wp = &Bsh[p][tid * 24];
      *(v8h*)wp       = blo;               // ds_store_b128 x2
      *(v8h*)(wp + 8) = bhi;
    }

    // ---- issue NEXT K-step's weight + scale loads before the barrier ----
    float nsc = 0.f, nzb = 0.f;
    if (ks + 1 < NSTEPS) {
      load_q16(qk, k0 + KSTEP, half, wn, qreg);
      const int gidx = kslice * GROUPS_PER_SLICE + ((ks + 1) >> 2);
      nsc = qs[(size_t)gidx * OUT_F + wn];
      nzb = qb[(size_t)gidx * OUT_F + wn];
    }

    // ---- issue this step's A-tile f32 loads (no LDS dependency) ----
    v4f u[2][4];
#pragma unroll
    for (int mt = 0; mt < 2; ++mt) {
      const float* rp = x + (size_t)(wave * 32 + mt * 16 + l16) * IN_F + k0 + 16 * half;
      u[mt][0] = *(const v4f*)(rp);
      u[mt][1] = *(const v4f*)(rp + 4);
      u[mt][2] = *(const v4f*)(rp + 8);
      u[mt][3] = *(const v4f*)(rp + 12);
    }

    __syncthreads();   // single barrier per K-step (double-buffered handoff)

    // ---- A fragments: cvt f32 -> f16 ----
    // A 16x32 f16 fragment: lane (m=l16) holds k = 16*half+0..15 (contiguous, same order as B)
    v16h afrag[2];
#pragma unroll
    for (int mt = 0; mt < 2; ++mt) {
      v16h a;
#pragma unroll
      for (int c = 0; c < 4; ++c) {
        a[4 * c + 0] = (_Float16)u[mt][c].x;
        a[4 * c + 1] = (_Float16)u[mt][c].y;
        a[4 * c + 2] = (_Float16)u[mt][c].z;
        a[4 * c + 3] = (_Float16)u[mt][c].w;
      }
      afrag[mt] = a;
    }

    // ---- 16 WMMAs: 8 n-tiles x 2 m-tiles, B from LDS buffer p ----
#pragma unroll
    for (int nt = 0; nt < 8; ++nt) {
      const _Float16* bp = &Bsh[p][(nt * 32 + lane) * 24];
      const v8h b0 = *(const v8h*)(bp);        // ds_load_b128
      const v8h b1 = *(const v8h*)(bp + 8);    // ds_load_b128
      const v16h bfrag = __builtin_shufflevector(
          b0, b1, 0, 1, 2, 3, 4, 5, 6, 7, 8, 9, 10, 11, 12, 13, 14, 15);
      acc[0][nt] = __builtin_amdgcn_wmma_f32_16x16x32_f16(
          false, afrag[0], false, bfrag, (short)0, acc[0][nt], false, false);
      acc[1][nt] = __builtin_amdgcn_wmma_f32_16x16x32_f16(
          false, afrag[1], false, bfrag, (short)0, acc[1][nt], false, false);
    }

    // ---- dequantize the next step's weights (loads have had the whole
    //      compute phase to land) ----
    if (ks + 1 < NSTEPS) dequant16(qreg, nsc, nzb, blo, bhi);
  }

  // ---- epilogue: split-K partials accumulate atomically (bias pre-stored) ----
  // C layout: VGPR r -> M = r + 8*(lane/16), N = lane%16
#pragma unroll
  for (int mt = 0; mt < 2; ++mt)
#pragma unroll
    for (int nt = 0; nt < 8; ++nt) {
      float* op = out + (size_t)(wave * 32 + mt * 16 + 8 * half) * OUT_F
                      + (n0 + nt * 16 + l16);
      const v8f a = acc[mt][nt];
#pragma unroll
      for (int r = 0; r < 8; ++r)
        atomicAdd(op + (size_t)r * OUT_F, a[r]);   // global_atomic_add_f32
    }
}

extern "C" void kernel_launch(void* const* d_in, const int* in_sizes, int n_in,
                              void* d_out, int out_size, void* d_ws, size_t ws_size,
                              hipStream_t stream) {
  const float* x    = (const float*)d_in[0];  // (8,32,4096) f32
  const int*   qk   = (const int*)  d_in[1];  // (4096,14336) int32
  const float* qs   = (const float*)d_in[2];  // (32,14336) f32
  const float* qb   = (const float*)d_in[3];  // (32,14336) f32
  const float* bias = (const float*)d_in[4];  // (14336,) f32
  float* out = (float*)d_out;                 // (8,32,14336) f32

  qlinear_init_bias<<<(M_TOT * OUT_F) / 256, 256, 0, stream>>>(bias, out);

  dim3 grid(OUT_F / BN, KSLICES);             // (112, 4)
  qlinear_wmma<<<grid, 256, 0, stream>>>(x, qk, qs, qb, out);
}
